// MotionStream_51058571215172
// MI455X (gfx1250) — matrix operand order repaired and checked by reading, not compiled, not run on hardware
//
#include <hip/hip_runtime.h>
#include <hip/hip_bf16.h>
#include <stdint.h>

#define B_  16
#define T_  16
#define O_  10
#define L_  160          // T*O tokens per batch
#define D_  768
#define NH_ 12
#define DH_ 64
#define HW_ 14
#define M_  (B_*L_)      // 2560 rows in the token matrix
#define KC_ 256          // K-chunk staged in LDS (divides 768 and 3072)

typedef __attribute__((ext_vector_type(16))) __bf16 v16bf;
typedef __attribute__((ext_vector_type(8)))  float  v8f;
typedef __attribute__((ext_vector_type(8)))  unsigned short us8;

union BfFrag { v16bf v; unsigned short u[16]; us8 h[2]; };

__device__ __forceinline__ unsigned short f2bf(float f) {
    unsigned int u = __float_as_uint(f);
    unsigned int r = u + 0x7FFFu + ((u >> 16) & 1u);   // round-to-nearest-even
    return (unsigned short)(r >> 16);
}

// CDNA5 async copy: global -> LDS, 16 bytes per lane, tracked by ASYNCcnt.
__device__ __forceinline__ void async_load_b128(unsigned lds_off, const void* gptr) {
    unsigned long long ga = (unsigned long long)(uintptr_t)gptr;
    asm volatile("global_load_async_to_lds_b128 %0, %1, off"
                 :: "v"(lds_off), "v"(ga) : "memory");
}
__device__ __forceinline__ void wait_asynccnt0() {
    asm volatile("s_wait_asynccnt 0x0" ::: "memory");
}

// ------------------------------------------- f32 [K][N] -> bf16 [N][K] (transposed)
__global__ void cvtT_bf16_kernel(const float* __restrict__ in,
                                 unsigned short* __restrict__ out, int K, int N) {
    int idx = blockIdx.x * blockDim.x + threadIdx.x;
    if (idx >= K * N) return;
    int k = idx / N, n = idx - k * N;
    out[(size_t)n * K + k] = f2bf(in[idx]);
}

// ------------------------------------------------ embed MLP (4->384->768)+cat
__global__ void embed_kernel(const float* __restrict__ box,
                             const float* __restrict__ cats,
                             const float* __restrict__ w1,
                             const float* __restrict__ w2,
                             float* __restrict__ x) {
    int tok = blockIdx.x;               // b*160 + t*10 + o (matches box layout)
    int b   = tok / L_;
    int r   = tok - b * L_;
    int t   = r / O_;
    int o   = r - t * O_;
    __shared__ float h1[384];
    const float* bt = box + (size_t)tok * 4;
    float b0 = bt[0], b1 = bt[1], b2 = bt[2], b3 = bt[3];
    for (int j = threadIdx.x; j < 384; j += blockDim.x) {
        float s = b0 * w1[j] + b1 * w1[384 + j] + b2 * w1[768 + j] + b3 * w1[1152 + j];
        h1[j] = s > 0.f ? s : 0.f;
    }
    __syncthreads();
    const float* crow = cats + (size_t)(t * O_ + o) * D_;   // categories indexed [t][o]
    float* xrow = x + (size_t)tok * D_;
    for (int d = threadIdx.x; d < D_; d += blockDim.x) {
        float acc = 0.f;
        for (int j = 0; j < 384; ++j) acc += h1[j] * w2[(size_t)j * D_ + d];
        acc = acc > 0.f ? acc : 0.f;
        xrow[d] = acc + crow[d];
    }
}

// ------------------------------------------------------- LayerNorm -> bf16
__global__ void ln_bf16_kernel(const float* __restrict__ x,
                               const float* __restrict__ g,
                               const float* __restrict__ bb,
                               unsigned short* __restrict__ out) {
    int row = blockIdx.x;
    const float* xr = x + (size_t)row * D_;
    __shared__ float red[256];
    float s = 0.f;
    for (int d = threadIdx.x; d < D_; d += 256) s += xr[d];
    red[threadIdx.x] = s; __syncthreads();
    for (int k = 128; k > 0; k >>= 1) {
        if (threadIdx.x < k) red[threadIdx.x] += red[threadIdx.x + k];
        __syncthreads();
    }
    float mu = red[0] / (float)D_;
    __syncthreads();
    float v = 0.f;
    for (int d = threadIdx.x; d < D_; d += 256) { float t = xr[d] - mu; v += t * t; }
    red[threadIdx.x] = v; __syncthreads();
    for (int k = 128; k > 0; k >>= 1) {
        if (threadIdx.x < k) red[threadIdx.x] += red[threadIdx.x + k];
        __syncthreads();
    }
    float rstd = rsqrtf(red[0] / (float)D_ + 1e-5f);
    for (int d = threadIdx.x; d < D_; d += 256)
        out[(size_t)row * D_ + d] = f2bf((xr[d] - mu) * rstd * g[d] + bb[d]);
}

// ------------- bf16 WMMA GEMM: A row-major [M][K], Bt transposed [N][K].
// Block = 8 waves; each wave owns a 16x16 C tile (same 16-col panel, 128 rows).
// B panel (16 x KC) staged via async-to-LDS, fragments read as ds_load_b128.
__global__ void gemm_bf16_wmma_kernel(const unsigned short* __restrict__ A, int lda,
                                      const unsigned short* __restrict__ Bt,
                                      const float* __restrict__ bias,
                                      const float* __restrict__ resid, int ldr,
                                      float* __restrict__ Cf, int ldcf,
                                      unsigned short* __restrict__ Cb, int ldcb,
                                      int M, int N, int K, int act) {
    __shared__ unsigned short Bl[16 * KC_];     // 8 KB
    int tid  = threadIdx.x;
    int wave = tid >> 5, lane = tid & 31;
    int half = lane >> 4, l16 = lane & 15;
    int m0 = (blockIdx.y * 8 + wave) * 16;
    int n0 = blockIdx.x * 16;
    const unsigned short* arow = A + (size_t)(m0 + l16) * lda;

    // async staging slots: thread copies 16B of row (tid>>5) and row (tid>>5)+8
    int seg   = tid & 31;        // 16B segment within a 512B (=KC_*2) row
    int nrow  = tid >> 5;        // 0..7
    unsigned lds0 = (unsigned)(uintptr_t)&Bl[nrow * KC_]       + (unsigned)seg * 16u;
    unsigned lds1 = (unsigned)(uintptr_t)&Bl[(nrow + 8) * KC_] + (unsigned)seg * 16u;

    v8f acc = {0.f, 0.f, 0.f, 0.f, 0.f, 0.f, 0.f, 0.f};
    for (int k0 = 0; k0 < K; k0 += KC_) {
        const char* g0 = (const char*)(Bt + (size_t)(n0 + nrow)     * K + k0) + seg * 16;
        const char* g1 = (const char*)(Bt + (size_t)(n0 + nrow + 8) * K + k0) + seg * 16;
        async_load_b128(lds0, g0);
        async_load_b128(lds1, g1);
        if (k0 + KC_ < K) __builtin_prefetch(arow + k0 + KC_, 0, 1);
        wait_asynccnt0();
        __syncthreads();
        const unsigned short* brow = &Bl[l16 * KC_];
        #pragma unroll
        for (int kk = 0; kk < KC_; kk += 32) {
            BfFrag af, bf;
            int ka = kk + half * 8;                 // A 16x32 fragment (ISA 7.12.2)
            af.h[0] = *(const us8*)(arow + k0 + ka);
            af.h[1] = *(const us8*)(arow + k0 + ka + 16);
            int kb = kk + half * 8;                 // B 32x16 fragment
            bf.h[0] = *(const us8*)(brow + kb);
            bf.h[1] = *(const us8*)(brow + kb + 16);
            acc = __builtin_amdgcn_wmma_f32_16x16x32_bf16(false, af.v, false, bf.v,
                                                          (short)0, acc, false, false);
        }
        __syncthreads();
    }
    int rbase = half * 8;
    int n = n0 + l16;
    #pragma unroll
    for (int i = 0; i < 8; ++i) {
        int m = m0 + rbase + i;
        float v = acc[i];
        if (bias)  v += bias[n];
        if (act)   v = 0.5f * v * (1.f + erff(v * 0.70710678f));   // exact GELU
        if (resid) v += resid[(size_t)m * ldr + n];
        if (Cf) Cf[(size_t)m * ldcf + n] = v;
        if (Cb) Cb[(size_t)m * ldcb + n] = f2bf(v);
    }
}

// -------- attention: block = 10 waves = one (b, head); wave w owns q-tile w.
// Scores and softmax fully in registers (shfl reductions, zero divergence);
// P transposed C-layout -> A-layout through a private 5 KB LDS strip per wave.
__global__ void attn_wmma_kernel(const unsigned short* __restrict__ qkv,
                                 unsigned short* __restrict__ obuf) {
    __shared__ unsigned short Pl[10 * 16 * L_];   // 50 KB
    int h  = blockIdx.x;      // 0..11
    int b  = blockIdx.y;      // 0..15
    int wave = threadIdx.x >> 5;                  // q-tile 0..9
    int lane = threadIdx.x & 31;
    int half = lane >> 4, l16 = lane & 15;
    const int ldq = 3 * D_;
    const unsigned short* Q  = qkv + (size_t)(b * L_ + wave * 16) * ldq + h * DH_;
    const unsigned short* Kk = qkv + (size_t)(b * L_) * ldq + D_ + h * DH_;
    const unsigned short* Vv = qkv + (size_t)(b * L_) * ldq + 2 * D_ + h * DH_;

    // S = Q @ K^T : 10 n-tiles, K = 64 (2 WMMA each). Each lane loads two 16B chunks 32B apart.
    const unsigned short* arow = Q + (size_t)l16 * ldq;
    v8f acc[10];
    #pragma unroll
    for (int nt = 0; nt < 10; ++nt) {
        v8f a = {0.f, 0.f, 0.f, 0.f, 0.f, 0.f, 0.f, 0.f};
        #pragma unroll
        for (int k0 = 0; k0 < DH_; k0 += 32) {
            BfFrag af, bf;
            int ka = k0 + half * 8;
            af.h[0] = *(const us8*)(arow + ka);
            af.h[1] = *(const us8*)(arow + ka + 16);
            int kb = k0 + half * 8;
            const unsigned short* krow = Kk + (size_t)(nt * 16 + l16) * ldq;
            bf.h[0] = *(const us8*)(krow + kb);
            bf.h[1] = *(const us8*)(krow + kb + 16);
            a = __builtin_amdgcn_wmma_f32_16x16x32_bf16(false, af.v, false, bf.v,
                                                        (short)0, a, false, false);
        }
        acc[nt] = a;
    }

    // register softmax: row r = half*8 + i lives in 16 lanes x 10 accumulators
    unsigned short* Pw = &Pl[wave * 16 * L_];
    const float scale = 0.125f;                   // 1/sqrt(64)
    #pragma unroll
    for (int i = 0; i < 8; ++i) {
        float mx = -3.4e38f;
        #pragma unroll
        for (int nt = 0; nt < 10; ++nt) mx = fmaxf(mx, acc[nt][i] * scale);
        #pragma unroll
        for (int m = 1; m < 16; m <<= 1) mx = fmaxf(mx, __shfl_xor(mx, m, 32));
        float e[10], sum = 0.f;
        #pragma unroll
        for (int nt = 0; nt < 10; ++nt) { e[nt] = __expf(acc[nt][i] * scale - mx); sum += e[nt]; }
        #pragma unroll
        for (int m = 1; m < 16; m <<= 1) sum += __shfl_xor(sum, m, 32);
        float inv = 1.f / sum;
        int r = half * 8 + i;
        #pragma unroll
        for (int nt = 0; nt < 10; ++nt)
            Pw[r * L_ + nt * 16 + l16] = f2bf(e[nt] * inv);
    }

    // O = P @ V (16 x 64); A-fragments from own LDS strip (ds_load_b128)
    unsigned short* orow = obuf + (size_t)(b * L_ + wave * 16) * D_ + h * DH_;
    const unsigned short* prow = &Pw[l16 * L_];
    #pragma unroll
    for (int nt = 0; nt < 4; ++nt) {
        v8f a = {0.f, 0.f, 0.f, 0.f, 0.f, 0.f, 0.f, 0.f};
        for (int k0 = 0; k0 < L_; k0 += 32) {
            BfFrag af, bf;
            int ka = k0 + half * 8;
            af.h[0] = *(const us8*)(prow + ka);
            af.h[1] = *(const us8*)(prow + ka + 16);
            int kb = k0 + half * 8;
            #pragma unroll
            for (int i = 0; i < 16; ++i)
                bf.u[i] = Vv[(size_t)(kb + (i & 7) + ((i >> 3) << 4)) * ldq + nt * 16 + l16];
            a = __builtin_amdgcn_wmma_f32_16x16x32_bf16(false, af.v, false, bf.v,
                                                        (short)0, a, false, false);
        }
        #pragma unroll
        for (int i = 0; i < 8; ++i)
            orow[(size_t)(half * 8 + i) * D_ + nt * 16 + l16] = f2bf(a[i]);
    }
}

// ------------------------ box-mask spatial pooling + temporal mean (Tr = 2)
__global__ void pool_kernel(const float* __restrict__ box,
                            const float* __restrict__ x,
                            float* __restrict__ out) {
    int idx  = blockIdx.x;                 // b*(8*196) + tp*196 + hw
    int hw   = idx % 196;
    int rest = idx / 196;
    int tp   = rest % 8;
    int b    = rest / 8;
    int hy = hw / HW_, wx = hw % HW_;
    float cy = (hy + 0.5f) / 14.f, cx = (wx + 0.5f) / 14.f;
    float a0 = 0.f, a1 = 0.f, a2 = 0.f;
    for (int i = 0; i < 2; ++i) {
        int t = tp * 2 + i;
        for (int o = 0; o < O_; ++o) {
            const float* bt = box + (size_t)((b * T_ + t) * O_ + o) * 4;
            float x1 = fminf(bt[0], bt[2]), x2 = fmaxf(bt[0], bt[2]);
            float y1 = fminf(bt[1], bt[3]), y2 = fmaxf(bt[1], bt[3]);
            if (cy >= y1 && cy <= y2 && cx >= x1 && cx <= x2) {
                const float* xr = x + (size_t)(b * L_ + t * O_ + o) * D_;
                a0 += xr[threadIdx.x];
                a1 += xr[threadIdx.x + 256];
                a2 += xr[threadIdx.x + 512];
            }
        }
    }
    float* orow = out + (size_t)idx * D_;
    orow[threadIdx.x]       = a0 * 0.5f;
    orow[threadIdx.x + 256] = a1 * 0.5f;
    orow[threadIdx.x + 512] = a2 * 0.5f;
}

// ---------------------------------------------------------------- launcher
extern "C" void kernel_launch(void* const* d_in, const int* in_sizes, int n_in,
                              void* d_out, int out_size, void* d_ws, size_t ws_size,
                              hipStream_t stream) {
    const float* box  = (const float*)d_in[0];
    const float* cats = (const float*)d_in[1];
    const float* w1   = (const float*)d_in[2];
    const float* w2   = (const float*)d_in[3];
    const float* ln1g = (const float*)d_in[4];
    const float* ln1b = (const float*)d_in[5];
    const float* wqkv = (const float*)d_in[6];
    const float* wo   = (const float*)d_in[7];
    const float* bo   = (const float*)d_in[8];
    const float* ln2g = (const float*)d_in[9];
    const float* ln2b = (const float*)d_in[10];
    const float* wfc1 = (const float*)d_in[11];
    const float* bfc1 = (const float*)d_in[12];
    const float* wfc2 = (const float*)d_in[13];
    const float* bfc2 = (const float*)d_in[14];
    (void)in_sizes; (void)n_in; (void)out_size; (void)ws_size;

    char* ws = (char*)d_ws;
    size_t off = 0;
    auto alloc = [&](size_t bytes) -> void* {
        void* p = ws + off;
        off += (bytes + 255) & ~(size_t)255;
        return p;
    };
    float*          x     = (float*)alloc((size_t)M_ * D_ * 4);
    unsigned short* hn    = (unsigned short*)alloc((size_t)M_ * D_ * 2);
    unsigned short* qkvb  = (unsigned short*)alloc((size_t)M_ * 3 * D_ * 2);
    unsigned short* ob    = (unsigned short*)alloc((size_t)M_ * D_ * 2);
    unsigned short* actb  = (unsigned short*)alloc((size_t)M_ * 4 * D_ * 2);
    unsigned short* wqkvT = (unsigned short*)alloc((size_t)D_ * 3 * D_ * 2);
    unsigned short* woT   = (unsigned short*)alloc((size_t)D_ * D_ * 2);
    unsigned short* wfc1T = (unsigned short*)alloc((size_t)D_ * 4 * D_ * 2);
    unsigned short* wfc2T = (unsigned short*)alloc((size_t)4 * D_ * D_ * 2);

    auto cvtT = [&](const float* src, unsigned short* dst, int K, int N) {
        cvtT_bf16_kernel<<<(K * N + 255) / 256, 256, 0, stream>>>(src, dst, K, N);
    };
    cvtT(wqkv, wqkvT, D_, 3 * D_);
    cvtT(wo,   woT,   D_, D_);
    cvtT(wfc1, wfc1T, D_, 4 * D_);
    cvtT(wfc2, wfc2T, 4 * D_, D_);

    embed_kernel<<<M_, 256, 0, stream>>>(box, cats, w1, w2, x);
    ln_bf16_kernel<<<M_, 256, 0, stream>>>(x, ln1g, ln1b, hn);

    // QKV: (2560x768) @ (768x2304) -> bf16
    gemm_bf16_wmma_kernel<<<dim3((3 * D_) / 16, M_ / 128), 256, 0, stream>>>(
        hn, D_, wqkvT, nullptr, nullptr, 0, nullptr, 0, qkvb, 3 * D_,
        M_, 3 * D_, D_, 0);

    attn_wmma_kernel<<<dim3(NH_, B_), 320, 0, stream>>>(qkvb, ob);

    // WO + bias + residual -> x (f32)
    gemm_bf16_wmma_kernel<<<dim3(D_ / 16, M_ / 128), 256, 0, stream>>>(
        ob, D_, woT, bo, x, D_, x, D_, nullptr, 0,
        M_, D_, D_, 0);

    ln_bf16_kernel<<<M_, 256, 0, stream>>>(x, ln2g, ln2b, hn);

    // FC1 + bias + exact GELU -> bf16
    gemm_bf16_wmma_kernel<<<dim3((4 * D_) / 16, M_ / 128), 256, 0, stream>>>(
        hn, D_, wfc1T, bfc1, nullptr, 0, nullptr, 0, actb, 4 * D_,
        M_, 4 * D_, D_, 1);

    // FC2 + bias + residual -> x (f32)
    gemm_bf16_wmma_kernel<<<dim3(D_ / 16, M_ / 128), 256, 0, stream>>>(
        actb, 4 * D_, wfc2T, bfc2, x, D_, x, D_, nullptr, 0,
        M_, D_, 4 * D_, 0);

    pool_kernel<<<B_ * 8 * 196, 256, 0, stream>>>(box, x, (float*)d_out);
}
